// BatchedACE_39384850104917
// MI455X (gfx1250) — matrix operand order repaired and checked by reading, not compiled, not run on hardware
//
#include <hip/hip_runtime.h>
#include <hip/hip_bf16.h>

typedef __attribute__((ext_vector_type(16))) _Float16 v16h;
typedef __attribute__((ext_vector_type(8)))  float    v8f;

#define NN 64      // M*B*H
#define TT 4096
#define DD 64
#define SS 128     // L_TAB * R
#define LL 8
#define RR 16
#define TC 128     // T-chunk per block
#define BLK 256
#define EPSV 1e-6f

// 16x32 f16 A-fragment pair K-index (ISA 7.12.2, 16-bit A layout)
__device__ __forceinline__ int a_kidx(int v, int half) {
    return (v >> 2) * 16 + half * 8 + (v & 3) * 2;
}
// 32x16 f16 B-fragment pair K-index
__device__ __forceinline__ int b_kidx(int v, int half) {
    return (v >> 2) * 16 + half * 8 + (v & 3) * 2;
}

// ---------------------------------------------------------------------------
// Kernel 1: proj(WMMA) -> softmax probsK -> WMMA accumulate b_sum, A
// ---------------------------------------------------------------------------
__global__ __launch_bounds__(BLK) void k1_bucket_accum(
    const float* __restrict__ Khf, const float* __restrict__ Vhf,
    const float* __restrict__ planes, const float* __restrict__ logit_temp,
    float* __restrict__ bsum, float* __restrict__ Acol)
{
    __shared__ _Float16 planesT[32 * DD];   //  4 KB [j][d]   (B-frag pairs contiguous)
    __shared__ _Float16 Kh[TC * DD];        // 16 KB [t][d]   (A-frag pairs contiguous)
    __shared__ float    proj_lds[TC * 32];  // 16 KB [t][j]
    __shared__ _Float16 probsT[SS * TC];    // 32 KB [s][t]   (A-frag pairs contiguous)
    __shared__ _Float16 Vt[DD * TC];        // 16 KB [d][t]   (B-frag pairs contiguous)

    const int n   = blockIdx.y;
    const int t0  = blockIdx.x * TC;
    const int tid = threadIdx.x;
    const int m   = n >> 4;                               // n / (B*H)
    const long base_nt = (long)(n >> 3) * TT * 512 + (long)(n & 7) * 64;

    for (int i = tid; i < 32 * DD; i += BLK) {            // coalesced read, transpose to [j][d]
        int d = i >> 5, j = i & 31;
        planesT[j * DD + d] = (_Float16)planes[m * DD * 32 + i];
    }
    for (int i = tid; i < TC * DD; i += BLK) {
        int t = i >> 6, d = i & 63;
        Kh[i] = (_Float16)Khf[base_nt + (long)(t0 + t) * 512 + d];
    }
    for (int i = tid; i < TC * DD; i += BLK) {
        int t = i >> 6, d = i & 63;
        Vt[d * TC + t] = (_Float16)Vhf[base_nt + (long)(t0 + t) * 512 + d];
    }
    __syncthreads();

    const int w = tid >> 5, lane = tid & 31, half = lane >> 4, row = lane & 15;

    // ---- projection GEMM: proj (TC x 32) = Kh (TC x 64) @ planes (64 x 32) ----
    // fixed trip count: every wave does exactly 2 tiles -> uniform, fully unrolled
    #pragma unroll
    for (int it = 0; it < 2; ++it) {
        int tile = w + it * 8;                // 0..15
        int ti = tile >> 1, tj = tile & 1;    // ti: t subtile 0..7, tj: j subtile 0..1
        v8f acc = {};
        #pragma unroll
        for (int kk = 0; kk < DD; kk += 32) {
            union { v16h h; unsigned int u[8]; } a, b;
            #pragma unroll
            for (int v = 0; v < 8; ++v) {
                a.u[v] = *(const unsigned int*)&Kh[(ti * 16 + row) * DD + kk + a_kidx(v, half)];
                b.u[v] = *(const unsigned int*)&planesT[(tj * 16 + row) * DD + kk + b_kidx(v, half)];
            }
            acc = __builtin_amdgcn_wmma_f32_16x16x32_f16(
                false, a.h, false, b.h, (short)0, acc, false, false);
        }
        #pragma unroll
        for (int i = 0; i < 8; ++i)
            proj_lds[(ti * 16 + half * 8 + i) * 32 + tj * 16 + row] = acc[i];
    }
    __syncthreads();

    float sc = expf(logit_temp[0]);
    sc = fminf(fmaxf(sc, 0.01f), 20.0f);
    const float inv = 1.0f / sc;

    // ---- tanh + corner logits + softmax over 16 buckets per table ----
    // logits[r]-max factors exactly: e[r] = prod_j (sign match ? 1 : exp(-2|th_j|))
    for (int pair = tid; pair < TC * LL; pair += BLK) {
        int t = pair >> 3, l = pair & 7;
        float th0 = tanhf(proj_lds[t * 32 + l * 4 + 0]) * inv;
        float th1 = tanhf(proj_lds[t * 32 + l * 4 + 1]) * inv;
        float th2 = tanhf(proj_lds[t * 32 + l * 4 + 2]) * inv;
        float th3 = tanhf(proj_lds[t * 32 + l * 4 + 3]) * inv;
        float g0 = expf(-2.0f * fabsf(th0));
        float g1 = expf(-2.0f * fabsf(th1));
        float g2 = expf(-2.0f * fabsf(th2));
        float g3 = expf(-2.0f * fabsf(th3));
        bool p0 = th0 >= 0.f, p1 = th1 >= 0.f, p2 = th2 >= 0.f, p3 = th3 >= 0.f;
        float e[RR]; float se = 0.f;
        #pragma unroll
        for (int r = 0; r < RR; ++r) {
            float f0 = (((r & 8) != 0) == p0) ? 1.f : g0;
            float f1 = (((r & 4) != 0) == p1) ? 1.f : g1;
            float f2 = (((r & 2) != 0) == p2) ? 1.f : g2;
            float f3 = (((r & 1) != 0) == p3) ? 1.f : g3;
            e[r] = (f0 * f1) * (f2 * f3);
            se += e[r];
        }
        float rse = 1.0f / se;
        #pragma unroll
        for (int r = 0; r < RR; ++r)
            probsT[(l * RR + r) * TC + t] = (_Float16)(e[r] * rse);
    }
    __syncthreads();

    // ---- A[s] partial column sums ----
    if (tid < SS) {
        float s = 0.f;
        for (int t = 0; t < TC; ++t) s += (float)probsT[tid * TC + t];
        atomicAdd(&Acol[n * SS + tid], s);
    }

    // ---- b_sum tile = probsK^T (S x TC) @ V (TC x D) ----
    #pragma unroll
    for (int it = 0; it < 4; ++it) {
        int tile = w + it * 8;                // 0..31
        int ti = tile >> 2, tj = tile & 3;    // ti: S subtile 0..7, tj: D subtile 0..3
        v8f acc = {};
        #pragma unroll
        for (int kk = 0; kk < TC; kk += 32) {
            union { v16h h; unsigned int u[8]; } a, b;
            #pragma unroll
            for (int v = 0; v < 8; ++v) {
                a.u[v] = *(const unsigned int*)&probsT[(ti * 16 + row) * TC + kk + a_kidx(v, half)];
                b.u[v] = *(const unsigned int*)&Vt[(tj * 16 + row) * TC + kk + b_kidx(v, half)];
            }
            acc = __builtin_amdgcn_wmma_f32_16x16x32_f16(
                false, a.h, false, b.h, (short)0, acc, false, false);
        }
        #pragma unroll
        for (int i = 0; i < 8; ++i) {
            int sg = ti * 16 + half * 8 + i;
            int dg = tj * 16 + row;
            atomicAdd(&bsum[((long)n * SS + sg) * DD + dg], acc[i]);
        }
    }
}

// ---------------------------------------------------------------------------
// Kernel 2: E = b_sum / (A + eps)  -> f16
// ---------------------------------------------------------------------------
__global__ __launch_bounds__(BLK) void k2_normalize(
    const float* __restrict__ bsum, const float* __restrict__ Acol,
    _Float16* __restrict__ E)
{
    int idx = blockIdx.x * BLK + threadIdx.x;
    if (idx >= NN * SS * DD) return;
    int n = idx >> 13;              // / (S*D)
    int s = (idx >> 6) & (SS - 1);
    E[idx] = (_Float16)(bsum[idx] / (Acol[n * SS + s] + EPSV));
}

// ---------------------------------------------------------------------------
// Kernel 3: proj(WMMA) -> softmax probsQ -> out = probsQ (TC x S) @ E (S x D)
// ---------------------------------------------------------------------------
__global__ __launch_bounds__(BLK) void k3_readout(
    const float* __restrict__ Qhf, const float* __restrict__ planes,
    const float* __restrict__ logit_temp, const _Float16* __restrict__ E,
    float* __restrict__ out)
{
    __shared__ _Float16 planesT[32 * DD];   //  4 KB [j][d]
    __shared__ _Float16 Qh[TC * DD];        // 16 KB [t][d]
    __shared__ float    proj_lds[TC * 32];  // 16 KB [t][j]
    __shared__ _Float16 probs[TC * SS];     // 32 KB [t][s]
    __shared__ _Float16 Et[DD * SS];        // 16 KB [d][s]

    const int n   = blockIdx.y;
    const int t0  = blockIdx.x * TC;
    const int tid = threadIdx.x;
    const int m   = n >> 4;
    const long base_nt = (long)(n >> 3) * TT * 512 + (long)(n & 7) * 64;

    for (int i = tid; i < 32 * DD; i += BLK) {
        int d = i >> 5, j = i & 31;
        planesT[j * DD + d] = (_Float16)planes[m * DD * 32 + i];
    }
    for (int i = tid; i < TC * DD; i += BLK) {
        int t = i >> 6, d = i & 63;
        Qh[i] = (_Float16)Qhf[base_nt + (long)(t0 + t) * 512 + d];
    }
    for (int i = tid; i < SS * DD; i += BLK) {
        int s = i >> 6, d = i & 63;
        Et[d * SS + s] = E[((long)n * SS + s) * DD + d];
    }
    __syncthreads();

    const int w = tid >> 5, lane = tid & 31, half = lane >> 4, row = lane & 15;

    // ---- projection GEMM ----
    #pragma unroll
    for (int it = 0; it < 2; ++it) {
        int tile = w + it * 8;
        int ti = tile >> 1, tj = tile & 1;
        v8f acc = {};
        #pragma unroll
        for (int kk = 0; kk < DD; kk += 32) {
            union { v16h h; unsigned int u[8]; } a, b;
            #pragma unroll
            for (int v = 0; v < 8; ++v) {
                a.u[v] = *(const unsigned int*)&Qh[(ti * 16 + row) * DD + kk + a_kidx(v, half)];
                b.u[v] = *(const unsigned int*)&planesT[(tj * 16 + row) * DD + kk + b_kidx(v, half)];
            }
            acc = __builtin_amdgcn_wmma_f32_16x16x32_f16(
                false, a.h, false, b.h, (short)0, acc, false, false);
        }
        #pragma unroll
        for (int i = 0; i < 8; ++i)
            proj_lds[(ti * 16 + half * 8 + i) * 32 + tj * 16 + row] = acc[i];
    }
    __syncthreads();

    float sc = expf(logit_temp[0]);
    sc = fminf(fmaxf(sc, 0.01f), 20.0f);
    const float inv = 1.0f / sc;

    for (int pair = tid; pair < TC * LL; pair += BLK) {
        int t = pair >> 3, l = pair & 7;
        float th0 = tanhf(proj_lds[t * 32 + l * 4 + 0]) * inv;
        float th1 = tanhf(proj_lds[t * 32 + l * 4 + 1]) * inv;
        float th2 = tanhf(proj_lds[t * 32 + l * 4 + 2]) * inv;
        float th3 = tanhf(proj_lds[t * 32 + l * 4 + 3]) * inv;
        float g0 = expf(-2.0f * fabsf(th0));
        float g1 = expf(-2.0f * fabsf(th1));
        float g2 = expf(-2.0f * fabsf(th2));
        float g3 = expf(-2.0f * fabsf(th3));
        bool p0 = th0 >= 0.f, p1 = th1 >= 0.f, p2 = th2 >= 0.f, p3 = th3 >= 0.f;
        float e[RR]; float se = 0.f;
        #pragma unroll
        for (int r = 0; r < RR; ++r) {
            float f0 = (((r & 8) != 0) == p0) ? 1.f : g0;
            float f1 = (((r & 4) != 0) == p1) ? 1.f : g1;
            float f2 = (((r & 2) != 0) == p2) ? 1.f : g2;
            float f3 = (((r & 1) != 0) == p3) ? 1.f : g3;
            e[r] = (f0 * f1) * (f2 * f3);
            se += e[r];
        }
        float rse = 1.0f / se;
        #pragma unroll
        for (int r = 0; r < RR; ++r)
            probs[t * SS + l * RR + r] = (_Float16)(e[r] * rse);
    }
    __syncthreads();

    // ---- readout GEMM: out chunk = probsQ (TC x S) @ E (S x D) ----
    #pragma unroll
    for (int it = 0; it < 4; ++it) {
        int tile = w + it * 8;
        int ti = tile >> 2, tj = tile & 3;
        v8f acc = {};
        #pragma unroll
        for (int kk = 0; kk < SS; kk += 32) {
            union { v16h h; unsigned int u[8]; } a, b;
            #pragma unroll
            for (int v = 0; v < 8; ++v) {
                a.u[v] = *(const unsigned int*)&probs[(ti * 16 + row) * SS + kk + a_kidx(v, half)];
                b.u[v] = *(const unsigned int*)&Et[(tj * 16 + row) * SS + kk + b_kidx(v, half)];
            }
            acc = __builtin_amdgcn_wmma_f32_16x16x32_f16(
                false, a.h, false, b.h, (short)0, acc, false, false);
        }
        #pragma unroll
        for (int i = 0; i < 8; ++i) {
            int tg = t0 + ti * 16 + half * 8 + i;
            int dg = tj * 16 + row;
            out[((long)n * TT + tg) * DD + dg] = acc[i];
        }
    }
}

// ---------------------------------------------------------------------------
extern "C" void kernel_launch(void* const* d_in, const int* in_sizes, int n_in,
                              void* d_out, int out_size, void* d_ws, size_t ws_size,
                              hipStream_t stream) {
    const float* Khf    = (const float*)d_in[0];
    const float* Vhf    = (const float*)d_in[1];
    const float* Qhf    = (const float*)d_in[2];
    const float* planes = (const float*)d_in[3];
    const float* lt     = (const float*)d_in[4];
    float* out = (float*)d_out;

    float*    bsum = (float*)d_ws;                     // N*S*D f32  (2 MB)
    float*    Acol = bsum + (size_t)NN * SS * DD;      // N*S  f32   (32 KB)
    _Float16* E    = (_Float16*)(Acol + NN * SS);      // N*S*D f16  (1 MB)

    // zero the atomic accumulators (harness poisons ws; must re-zero every call)
    hipMemsetAsync(d_ws, 0, ((size_t)NN * SS * DD + (size_t)NN * SS) * sizeof(float), stream);

    dim3 grid(TT / TC, NN);
    k1_bucket_accum<<<grid, BLK, 0, stream>>>(Khf, Vhf, planes, lt, bsum, Acol);
    k2_normalize<<<(NN * SS * DD + BLK - 1) / BLK, BLK, 0, stream>>>(bsum, Acol, E);
    k3_readout<<<grid, BLK, 0, stream>>>(Qhf, planes, lt, E, out);
}
